// CrossAttention_82944408420623
// MI455X (gfx1250) — matrix operand order repaired and checked by reading, not compiled, not run on hardware
//
#include <hip/hip_runtime.h>
#include <math.h>

// B=4, C=256, N=64*64=4096 cross-attention, flash-attention style, bf16 WMMA,
// with double-buffered async global->LDS staging of K/V blocks (ASYNCcnt path).

#define BB 4
#define CC 256
#define NN 4096

typedef __bf16 bf16;
typedef __attribute__((ext_vector_type(16))) __bf16 bf16x16;
typedef __attribute__((ext_vector_type(8)))  float  v8f;
typedef __attribute__((ext_vector_type(4)))  float  v4f;
typedef __attribute__((ext_vector_type(4)))  int    v4i;

union FragU { bf16x16 v; v4i i[2]; };
union PackU { v4i i; bf16 h[8]; };
union F4U   { v4f f; float s[4]; };

__device__ __forceinline__ bf16x16 ld_frag_g(const bf16* p0, const bf16* p1) {
  FragU u;
  u.i[0] = *(const v4i*)p0;
  u.i[1] = *(const v4i*)p1;
  return u.v;
}

__device__ __forceinline__ v8f wmma_bf16f32(bf16x16 a, bf16x16 b, v8f c) {
  return __builtin_amdgcn_wmma_f32_16x16x32_bf16(false, a, false, b, (short)0, c,
                                                 false, false);
}

// one 16-byte async global->LDS chunk; lds_addr = raw LDS byte offset (per lane)
__device__ __forceinline__ void async_b128(unsigned lds_addr, const void* g) {
  asm volatile("global_load_async_to_lds_b128 %0, %1, off"
               :: "v"(lds_addr), "v"((unsigned long long)(uintptr_t)g)
               : "memory");
}

// ---------------- fp32 [B][C][N] -> bf16 [B][N][C] (transpose + convert) ----
__global__ void __launch_bounds__(256)
k_transpose_cvt(const float* __restrict__ src, bf16* __restrict__ dst) {
  __shared__ float tile[32][33];
  int b = blockIdx.z;
  int n0 = blockIdx.x * 32;
  int c0 = blockIdx.y * 32;
  int tx = threadIdx.x, ty = threadIdx.y;
#pragma unroll
  for (int r = 0; r < 4; ++r) {
    int c = c0 + ty + r * 8;
    tile[ty + r * 8][tx] = src[((size_t)b * CC + c) * NN + n0 + tx];
  }
  __syncthreads();
#pragma unroll
  for (int r = 0; r < 4; ++r) {
    int n = n0 + ty + r * 8;
    dst[((size_t)b * NN + n) * CC + c0 + tx] = (bf16)tile[tx][ty + r * 8];
  }
}

// ---------------- weights fp32 -> bf16 (row-major, no transpose) ------------
__global__ void __launch_bounds__(256)
k_cvt_w(const float* __restrict__ wq, const float* __restrict__ wk,
        const float* __restrict__ wv, bf16* __restrict__ dq,
        bf16* __restrict__ dk, bf16* __restrict__ dv) {
  int idx = blockIdx.x * blockDim.x + threadIdx.x;
  int m = idx >> 16;   // 65536 elements per matrix
  int i = idx & 65535;
  const float* s = (m == 0) ? wq : ((m == 1) ? wk : wv);
  bf16* d = (m == 0) ? dq : ((m == 1) ? dk : dv);
  d[i] = (bf16)s[i];
}

// ---------------- QKV projection (WMMA bf16) --------------------------------
__global__ void __launch_bounds__(128)
k_proj(const bf16* __restrict__ tT, const bf16* __restrict__ rT,
       const bf16* __restrict__ Wq, const bf16* __restrict__ Wk,
       const bf16* __restrict__ Wv, const float* __restrict__ bq,
       const float* __restrict__ bk, const float* __restrict__ bv,
       bf16* __restrict__ Q, bf16* __restrict__ K, bf16* __restrict__ Vt) {
  int lane = threadIdx.x & 31;
  int w = threadIdx.x >> 5;
  int gw = blockIdx.x * 4 + w;            // 1024 waves total
  int b = gw >> 8;
  int n0 = (gw & 255) << 4;               // 16-pixel slab
  int h = lane >> 4, l16 = lane & 15;

  bf16x16 bfr[8];

  // ---- Q: D[o][n] = Wq x tT^T, stored transposed into Q[b][n][o] ----
#pragma unroll
  for (int cc = 0; cc < 8; ++cc) {
    const bf16* p = tT + ((size_t)b * NN + n0 + l16) * CC + cc * 32 + h * 8;
    bfr[cc] = ld_frag_g(p, p + 16);
  }
#pragma unroll 1
  for (int ot = 0; ot < 16; ++ot) {
    int o0 = ot * 16;
    v8f acc = {0.f, 0.f, 0.f, 0.f, 0.f, 0.f, 0.f, 0.f};
#pragma unroll
    for (int cc = 0; cc < 8; ++cc) {
      const bf16* p0 = Wq + (size_t)(o0 + l16) * CC + cc * 32 + h * 8;
      acc = wmma_bf16f32(ld_frag_g(p0, p0 + 16), bfr[cc], acc);
    }
    F4U b0, b1;
    b0.f = *(const v4f*)(bq + o0 + 8 * h);
    b1.f = *(const v4f*)(bq + o0 + 8 * h + 4);
    PackU st;
#pragma unroll
    for (int v = 0; v < 8; ++v)
      st.h[v] = (bf16)(acc[v] + (v < 4 ? b0.s[v] : b1.s[v - 4]));
    *(v4i*)(Q + ((size_t)b * NN + n0 + l16) * CC + o0 + 8 * h) = st.i;
  }

  // ---- K: same with reference / Wk / bk ----
#pragma unroll
  for (int cc = 0; cc < 8; ++cc) {
    const bf16* p = rT + ((size_t)b * NN + n0 + l16) * CC + cc * 32 + h * 8;
    bfr[cc] = ld_frag_g(p, p + 16);
  }
#pragma unroll 1
  for (int ot = 0; ot < 16; ++ot) {
    int o0 = ot * 16;
    v8f acc = {0.f, 0.f, 0.f, 0.f, 0.f, 0.f, 0.f, 0.f};
#pragma unroll
    for (int cc = 0; cc < 8; ++cc) {
      const bf16* p0 = Wk + (size_t)(o0 + l16) * CC + cc * 32 + h * 8;
      acc = wmma_bf16f32(ld_frag_g(p0, p0 + 16), bfr[cc], acc);
    }
    F4U b0, b1;
    b0.f = *(const v4f*)(bk + o0 + 8 * h);
    b1.f = *(const v4f*)(bk + o0 + 8 * h + 4);
    PackU st;
#pragma unroll
    for (int v = 0; v < 8; ++v)
      st.h[v] = (bf16)(acc[v] + (v < 4 ? b0.s[v] : b1.s[v - 4]));
    *(v4i*)(K + ((size_t)b * NN + n0 + l16) * CC + o0 + 8 * h) = st.i;
  }

  // ---- V: D[n][o] = rT x Wv^T, stored transposed into Vt[b][o][n] ----
  bf16x16 afr[8];
#pragma unroll
  for (int cc = 0; cc < 8; ++cc) {
    const bf16* p0 = rT + ((size_t)b * NN + n0 + l16) * CC + cc * 32 + h * 8;
    afr[cc] = ld_frag_g(p0, p0 + 16);
  }
#pragma unroll 1
  for (int ot = 0; ot < 16; ++ot) {
    int o0 = ot * 16;
    v8f acc = {0.f, 0.f, 0.f, 0.f, 0.f, 0.f, 0.f, 0.f};
#pragma unroll
    for (int cc = 0; cc < 8; ++cc) {
      const bf16* p = Wv + (size_t)(o0 + l16) * CC + cc * 32 + h * 8;
      acc = wmma_bf16f32(afr[cc], ld_frag_g(p, p + 16), acc);
    }
    float bb = bv[o0 + l16];
    PackU st;
#pragma unroll
    for (int v = 0; v < 8; ++v) st.h[v] = (bf16)(acc[v] + bb);
    *(v4i*)(Vt + ((size_t)b * CC + o0 + l16) * NN + n0 + 8 * h) = st.i;
  }
}

// ---------------- flash attention ------------------------------------------
// Block = 4 waves; each wave owns one 16-row i-block. K/V j-blocks are staged
// cooperatively into LDS with double-buffered async copies (ASYNCcnt).
__global__ void __launch_bounds__(128)
k_attn(const bf16* __restrict__ Qb, const bf16* __restrict__ Kb,
       const bf16* __restrict__ Vt, const float* __restrict__ tgt,
       float* __restrict__ out) {
  __shared__ bf16 ldsK[2][32 * 256];      // K rows j0..j0+31, [j][c], 16 KB each
  __shared__ bf16 ldsV[2][256 * 32];      // V window, [c][j], 16 KB each
  __shared__ bf16 ldsP[4][16 * 32];       // per-wave P tile

  int tid = threadIdx.x;
  int lane = tid & 31;
  int w = tid >> 5;
  int gw = blockIdx.x * 4 + w;            // 1024 waves total
  int b = gw >> 8;
  int i0 = (gw & 255) << 4;
  int h = lane >> 4, l16 = lane & 15;

  const bf16* Kbase = Kb + (size_t)b * NN * CC;
  const bf16* Vbase = Vt + (size_t)b * CC * NN;
  unsigned kLds0 = (unsigned)(uintptr_t)&ldsK[0][0];
  unsigned vLds0 = (unsigned)(uintptr_t)&ldsV[0][0];

  // issue 16 async b128 copies (this wave's quarter is implicit: all 128
  // threads participate; each thread moves 8 chunks of K + 8 of V)
  auto issue_copy = [&](int buf, int j0) {
    unsigned kL = kLds0 + (unsigned)buf * (32 * 256 * 2);
    const bf16* kG = Kbase + (size_t)j0 * CC;       // contiguous 16 KB
#pragma unroll
    for (int k = 0; k < 8; ++k) {
      unsigned e = (unsigned)(k * 128 + tid) * 8;   // element offset
      async_b128(kL + e * 2, kG + e);
    }
    unsigned vL = vLds0 + (unsigned)buf * (256 * 32 * 2);
#pragma unroll
    for (int k = 0; k < 8; ++k) {
      int q = k * 128 + tid;                        // 1024 chunks of 16 B
      int row = q >> 2, sub = q & 3;
      async_b128(vL + (unsigned)(row * 64 + sub * 16),
                 Vbase + (size_t)row * NN + j0 + sub * 8);
    }
  };

  issue_copy(0, 0);

  // Q row-block A-fragments (16 x 256), resident
  bf16x16 qf[8];
#pragma unroll
  for (int cc = 0; cc < 8; ++cc) {
    const bf16* p0 = Qb + ((size_t)b * NN + i0 + l16) * CC + cc * 32 + h * 8;
    qf[cc] = ld_frag_g(p0, p0 + 16);
  }

  v8f acc[16];
  const v8f vzero = {0.f, 0.f, 0.f, 0.f, 0.f, 0.f, 0.f, 0.f};
#pragma unroll
  for (int t = 0; t < 16; ++t) acc[t] = vzero;
  float mrow[8], lrow[8];
#pragma unroll
  for (int v = 0; v < 8; ++v) { mrow[v] = -3.0e38f; lrow[v] = 0.f; }

  for (int it = 0; it < NN / 32; ++it) {
    int buf = it & 1;
    // prefetch next j-block (wraps to 0 on last iter; harmless, keeps waits
    // branchless; S_ENDPGM's implicit wait-idle retires the dangling copy)
    issue_copy(buf ^ 1, ((it + 1) * 32) & (NN - 1));
    // in-order ASYNCcnt: <=16 outstanding means current buffer is complete
    asm volatile("s_wait_asynccnt 0x10" ::: "memory");
    __syncthreads();

    const bf16* ldk = &ldsK[buf][0];
    const bf16* ldv = &ldsV[buf][0];

    // S tile (16 x 32) = Q x K^T, fragments from LDS
    v8f s0 = vzero, s1 = vzero;
#pragma unroll
    for (int cc = 0; cc < 8; ++cc) {
      const bf16* p = ldk + (size_t)l16 * CC + cc * 32 + h * 8;
      s0 = wmma_bf16f32(qf[cc], ld_frag_g(p, p + 16), s0);
    }
#pragma unroll
    for (int cc = 0; cc < 8; ++cc) {
      const bf16* p = ldk + (size_t)(16 + l16) * CC + cc * 32 + h * 8;
      s1 = wmma_bf16f32(qf[cc], ld_frag_g(p, p + 16), s1);
    }

    // online softmax; row r = v + 8*h lives at VGPR slot v within lane-half h
    float alpha[8];
#pragma unroll
    for (int v = 0; v < 8; ++v) {
      float a0 = s0[v] * 0.0625f;          // scale = C^-0.5 = 1/16
      float a1 = s1[v] * 0.0625f;
      float t = fmaxf(a0, a1);
#pragma unroll
      for (int m = 1; m <= 8; m <<= 1) t = fmaxf(t, __shfl_xor(t, m, 32));
      float mn = fmaxf(mrow[v], t);
      alpha[v] = __expf(mrow[v] - mn);
      mrow[v] = mn;
      float p0 = __expf(a0 - mn);
      float p1 = __expf(a1 - mn);
      float rs = p0 + p1;
#pragma unroll
      for (int m = 1; m <= 8; m <<= 1) rs += __shfl_xor(rs, m, 32);
      lrow[v] = lrow[v] * alpha[v] + rs;
      ldsP[w][(v + 8 * h) * 32 + l16] = (bf16)p0;
      ldsP[w][(v + 8 * h) * 32 + 16 + l16] = (bf16)p1;
    }

    // rescale running output accumulators
#pragma unroll
    for (int t = 0; t < 16; ++t) {
#pragma unroll
      for (int v = 0; v < 8; ++v) acc[t][v] = acc[t][v] * alpha[v];
    }

    // per-wave LDS ops are in-order; fence stops compiler reordering
    asm volatile("s_wait_dscnt 0x0" ::: "memory");

    // reload P in A-fragment layout (lane=row, K runs per half)
    bf16x16 pf;
    {
      const bf16* lp = &ldsP[w][l16 * 32 + h * 8];
      FragU u;
      u.i[0] = *(const v4i*)lp;
      u.i[1] = *(const v4i*)(lp + 16);
      pf = u.v;
    }

    // O += P x V (V B-fragment: contiguous j-run in LDS [c][j] window)
#pragma unroll
    for (int ct = 0; ct < 16; ++ct) {
      const bf16* p = ldv + (size_t)(ct * 16 + l16) * 32 + h * 8;
      acc[ct] = wmma_bf16f32(pf, ld_frag_g(p, p + 16), acc[ct]);
    }

    __syncthreads();   // all waves done reading buf before it is overwritten
  }

  // epilogue: divide by row sums, add residual target, store fp32
  float rl[8];
#pragma unroll
  for (int v = 0; v < 8; ++v) rl[v] = 1.0f / lrow[v];
#pragma unroll
  for (int ct = 0; ct < 16; ++ct) {
    int c = ct * 16 + l16;
    size_t base = ((size_t)b * CC + c) * NN + i0 + 8 * h;
    F4U t0, t1, o0u, o1u;
    t0.f = *(const v4f*)(tgt + base);
    t1.f = *(const v4f*)(tgt + base + 4);
#pragma unroll
    for (int v = 0; v < 4; ++v) o0u.s[v] = acc[ct][v] * rl[v] + t0.s[v];
#pragma unroll
    for (int v = 0; v < 4; ++v) o1u.s[v] = acc[ct][v + 4] * rl[v + 4] + t1.s[v];
    *(v4f*)(out + base) = o0u.f;
    *(v4f*)(out + base + 4) = o1u.f;
  }
}

extern "C" void kernel_launch(void* const* d_in, const int* in_sizes, int n_in,
                              void* d_out, int out_size, void* d_ws,
                              size_t ws_size, hipStream_t stream) {
  const float* tgt = (const float*)d_in[0];
  const float* ref = (const float*)d_in[1];
  const float* Wq = (const float*)d_in[2];
  const float* bq = (const float*)d_in[3];
  const float* Wk = (const float*)d_in[4];
  const float* bk = (const float*)d_in[5];
  const float* Wv = (const float*)d_in[6];
  const float* bv = (const float*)d_in[7];
  float* out = (float*)d_out;

  char* ws = (char*)d_ws;
  const size_t szT = (size_t)BB * NN * CC * sizeof(bf16);  // 8 MB each
  bf16* tT = (bf16*)(ws + 0 * szT);
  bf16* rT = (bf16*)(ws + 1 * szT);
  bf16* Qb = (bf16*)(ws + 2 * szT);
  bf16* Kb = (bf16*)(ws + 3 * szT);
  bf16* Vt = (bf16*)(ws + 4 * szT);
  bf16* Wqb = (bf16*)(ws + 5 * szT);
  bf16* Wkb = Wqb + (size_t)CC * CC;
  bf16* Wvb = Wkb + (size_t)CC * CC;

  dim3 tgrid(NN / 32, CC / 32, BB), tblk(32, 8);
  k_transpose_cvt<<<tgrid, tblk, 0, stream>>>(tgt, tT);
  k_transpose_cvt<<<tgrid, tblk, 0, stream>>>(ref, rT);
  k_cvt_w<<<768, 256, 0, stream>>>(Wq, Wk, Wv, Wqb, Wkb, Wvb);
  k_proj<<<256, 128, 0, stream>>>(tT, rT, Wqb, Wkb, Wvb, bq, bk, bv, Qb, Kb, Vt);
  k_attn<<<256, 128, 0, stream>>>(Qb, Kb, Vt, tgt, out);
}
